// Model_55877524521151
// MI455X (gfx1250) — matrix operand order repaired and checked by reading, not compiled, not run on hardware
//
#include <hip/hip_runtime.h>
#include <math.h>

typedef _Float16 v16h __attribute__((ext_vector_type(16)));
typedef _Float16 h2   __attribute__((ext_vector_type(2)));
typedef float    v8f  __attribute__((ext_vector_type(8)));
typedef float    f4   __attribute__((ext_vector_type(4)));

constexpr int kS  = 1024;   // query length
constexpr int kB  = 4;      // batch
constexpr int kNH = 12;     // heads
constexpr int kDH = 16;     // head dim
constexpr int kDM = 192;    // model dim = 12*16
constexpr int kR  = 2048;   // rel length
constexpr int kRP = kR + 16;// padded rel rows (zero tail)
constexpr int kBN = kB * kNH; // 48
// softmax done in base-2 domain: scale = (1/sqrt(16)) * log2(e)
constexpr float kScale2 = 0.25f * 1.44269504088896340736f;

__device__ __forceinline__ v8f wmma_f16(v16h a, v16h b, v8f c) {
  // D = A(16x32 f16) x B(32x16 f16) + C(16x16 f32)
  return __builtin_amdgcn_wmma_f32_16x16x32_f16(false, a, false, b, (short)0, c,
                                                false, false);
}

// ---------------------------------------------------------------------------
// Projections from x: q (+both biases), k (row-major), v (transposed)
// grid (48, 16) x 128 threads; wave handles a 16-token tile, K=192 in 6 steps
// ---------------------------------------------------------------------------
__global__ __launch_bounds__(128)
void k_proj_x(const float* __restrict__ x,
              const float* __restrict__ Wq, const float* __restrict__ Wk,
              const float* __restrict__ Wv,
              const float* __restrict__ rwb, const float* __restrict__ rrb,
              _Float16* __restrict__ qw, _Float16* __restrict__ qr,
              _Float16* __restrict__ kmat, _Float16* __restrict__ vT) {
  __shared__ __align__(32) _Float16 WT[3][kDH][kDM];  // W transposed: [d][h]
  const int bn = blockIdx.x, b = bn / kNH, n = bn % kNH;
  const int tid = threadIdx.x;
  for (int idx = tid; idx < kDH * kDM; idx += 128) {
    int d = idx / kDM, h = idx % kDM;
    WT[0][d][h] = (_Float16)Wq[(h * kNH + n) * kDH + d];
    WT[1][d][h] = (_Float16)Wk[(h * kNH + n) * kDH + d];
    WT[2][d][h] = (_Float16)Wv[(h * kNH + n) * kDH + d];
  }
  __syncthreads();
  const int l = tid & 31;
  const int i0 = blockIdx.y * 64 + (tid >> 5) * 16;
  const int rowA = l & 15, kbA = (l >> 4) * 16;  // A: lane=M, K contiguous 16
  const int colB = l & 15, kbB = (l >> 4) * 16;  // B: lane=N, K contiguous 16
  v8f aq = {}, ak = {}, av = {};
  for (int kk = 0; kk < kDM; kk += 32) {
    v16h a = {};
    const float* xp = x + ((size_t)(i0 + rowA) * kB + b) * kDM + kk + kbA;
#pragma unroll
    for (int e = 0; e < 8; ++e) { a[e] = (_Float16)xp[e]; a[8 + e] = (_Float16)xp[8 + e]; }
    v16h bq = *(const v16h*)&WT[0][colB][kk + kbB];
    v16h bk = *(const v16h*)&WT[1][colB][kk + kbB];
    v16h bv = *(const v16h*)&WT[2][colB][kk + kbB];
    aq = wmma_f16(a, bq, aq);
    ak = wmma_f16(a, bk, ak);
    av = wmma_f16(a, bv, av);
  }
  const float bw = rwb[n * kDH + colB], br = rrb[n * kDH + colB];
#pragma unroll
  for (int v = 0; v < 8; ++v) {
    int M = v + 8 * (l >> 4);
    size_t i = (size_t)i0 + M;
    size_t o = ((size_t)bn * kS + i) * kDH + colB;
    qw[o]   = (_Float16)(aq[v] + bw);
    qr[o]   = (_Float16)(aq[v] + br);
    kmat[o] = (_Float16)ak[v];
    vT[((size_t)bn * kDH + colB) * kS + i] = (_Float16)av[v];  // transposed
  }
}

// ---------------------------------------------------------------------------
// Projection of r -> kr (row-major [bn][rel][16]); grid (48, 32) x 128
// ---------------------------------------------------------------------------
__global__ __launch_bounds__(128)
void k_proj_r(const float* __restrict__ r, const float* __restrict__ Wr,
              _Float16* __restrict__ kr) {
  __shared__ __align__(32) _Float16 WT[kDH][kDM];
  const int bn = blockIdx.x, b = bn / kNH, n = bn % kNH;
  const int tid = threadIdx.x;
  for (int idx = tid; idx < kDH * kDM; idx += 128) {
    int d = idx / kDM, h = idx % kDM;
    WT[d][h] = (_Float16)Wr[(h * kNH + n) * kDH + d];
  }
  __syncthreads();
  const int l = tid & 31;
  const int i0 = blockIdx.y * 64 + (tid >> 5) * 16;
  const int rowA = l & 15, kbA = (l >> 4) * 16;
  const int colB = l & 15, kbB = (l >> 4) * 16;
  v8f acc = {};
  for (int kk = 0; kk < kDM; kk += 32) {
    v16h a = {};
    const float* rp = r + ((size_t)(i0 + rowA) * kB + b) * kDM + kk + kbA;
#pragma unroll
    for (int e = 0; e < 8; ++e) { a[e] = (_Float16)rp[e]; a[8 + e] = (_Float16)rp[8 + e]; }
    v16h bw = *(const v16h*)&WT[colB][kk + kbB];
    acc = wmma_f16(a, bw, acc);
  }
#pragma unroll
  for (int v = 0; v < 8; ++v) {
    int M = v + 8 * (l >> 4);
    kr[((size_t)bn * kRP + i0 + M) * kDH + colB] = (_Float16)acc[v];
  }
}

// zero the 16 padded rel rows per (b,n) so the bd window never reads garbage
__global__ void k_pad_kr(_Float16* __restrict__ kr) {
  int idx = blockIdx.x * 256 + threadIdx.x;
  const int tot = kBN * 16 * kDH;
  if (idx < tot) {
    int bn = idx / (16 * kDH);
    int rem = idx % (16 * kDH);
    kr[((size_t)bn * kRP + kR) * kDH + rem] = (_Float16)0;
  }
}

// ---------------------------------------------------------------------------
// Fused flash attention with relative shift.
// One wave per (b, n, 16-row i-block); j processed in steps of 64.
// bd_shifted[i,j] = qr_i . kr_{j-i+S}  (RLEN = 2S identity)
// ---------------------------------------------------------------------------
__global__ __launch_bounds__(32)
void k_attn(const _Float16* __restrict__ qw, const _Float16* __restrict__ qr,
            const _Float16* __restrict__ kmat, const _Float16* __restrict__ kr,
            const _Float16* __restrict__ vT, _Float16* __restrict__ avec) {
  __shared__ __align__(16) float    ac_s[16][64];
  __shared__ __align__(16) float    bd_s[16][80];
  __shared__ __align__(32) _Float16 p_s[16][64];
  __shared__ float alpha_s[16];
  __shared__ float sum_s[16];

  const int bn = blockIdx.x % kBN;
  const int i0 = (blockIdx.x / kBN) * 16;
  const int b = bn / kNH, n = bn % kNH;
  const int l = threadIdx.x;
  const int rowA = l & 15, kbA = (l >> 4) * 8;
  const int half = l >> 4;

  // Q fragments: d_head=16 padded to K=32 with zeros (upper halves stay 0)
  v16h aQW = {}, aQR = {};
  {
    const _Float16* qwp = qw + ((size_t)bn * kS + i0 + rowA) * kDH + kbA;
    const _Float16* qrp = qr + ((size_t)bn * kS + i0 + rowA) * kDH + kbA;
#pragma unroll
    for (int e = 0; e < 16; ++e) if (half == 0) { aQW[e] = qwp[e]; aQR[e] = qrp[e]; }
  }

  const _Float16* kbase  = kmat + (size_t)bn * kS * kDH;
  const _Float16* krbase = kr   + (size_t)bn * kRP * kDH;
  const _Float16* vbase  = vT   + (size_t)bn * kDH * kS;

  v8f O = {};
  float m = -INFINITY, ssum = 0.0f;

  for (int j0 = 0; j0 < kS; j0 += 64) {
    // ---- content scores ac: 4 WMMA tiles (B lanes>=16 carry d-padding = 0)
#pragma unroll
    for (int t = 0; t < 4; ++t) {
      v16h bK = {};
      if (l < 16) bK = *(const v16h*)(kbase + (size_t)(j0 + t * 16 + l) * kDH);
      v8f c = {};
      c = wmma_f16(aQW, bK, c);
#pragma unroll
      for (int v = 0; v < 8; ++v) ac_s[v + 8 * half][t * 16 + rowA] = c[v];
    }
    // ---- position scores bd: 79-wide rel window -> 5 WMMA tiles
    const int base = j0 - i0 + kS - 15;   // in [1, 1969]; max row read 2048 < kRP
#pragma unroll
    for (int t = 0; t < 5; ++t) {
      v16h bR = {};
      if (l < 16) bR = *(const v16h*)(krbase + (size_t)(base + t * 16 + l) * kDH);
      v8f c = {};
      c = wmma_f16(aQR, bR, c);
#pragma unroll
      for (int v = 0; v < 8; ++v) bd_s[v + 8 * half][t * 16 + rowA] = c[v];
    }
    if (j0 + 64 < kS) {  // prefetch next step's K/Kr/V cachelines (128B/lane)
      if (l < 16) {
        __builtin_prefetch(kbase + (size_t)(j0 + 64) * kDH + l * 64, 0, 3);
        __builtin_prefetch(vbase + (size_t)l * kS + j0 + 64, 0, 3);
      }
      if (l < 20)
        __builtin_prefetch(krbase + (size_t)(base + 64) * kDH + l * 64, 0, 3);
    }
    __syncthreads();

    // ---- online softmax (base-2): lane owns row rowA, cols [half*32,+32)
    float sv[32];
    float smax = -INFINITY;
#pragma unroll
    for (int c4 = 0; c4 < 8; ++c4) {
      f4 a4 = *(const f4*)&ac_s[rowA][half * 32 + c4 * 4];
#pragma unroll
      for (int e = 0; e < 4; ++e) {
        int c = c4 * 4 + e;
        int col = half * 32 + c;
        float s = (a4[e] + bd_s[rowA][col - rowA + 15]) * kScale2;
        sv[c] = s;
        smax = fmaxf(smax, s);
      }
    }
    smax = fmaxf(smax, __shfl_xor(smax, 16, 32));
    float mnew  = fmaxf(m, smax);
    float alpha = exp2f(m - mnew);
    float psum = 0.0f;
#pragma unroll
    for (int c = 0; c < 32; c += 2) {
      float p0 = exp2f(sv[c] - mnew);
      float p1 = exp2f(sv[c + 1] - mnew);
      psum += p0 + p1;
      h2 pp = { (_Float16)p0, (_Float16)p1 };
      *(h2*)&p_s[rowA][half * 32 + c] = pp;
    }
    psum += __shfl_xor(psum, 16, 32);
    ssum = ssum * alpha + psum;
    m = mnew;
    if (half == 0) alpha_s[rowA] = alpha;
    __syncthreads();

    // ---- rescale O (C layout: lane row M = v + 8*half), accumulate P x V
#pragma unroll
    for (int v = 0; v < 8; ++v) O[v] *= alpha_s[v + 8 * half];
#pragma unroll
    for (int js = 0; js < 64; js += 32) {
      v16h aP;
#pragma unroll
      for (int e = 0; e < 8; ++e) {
        aP[e]     = p_s[rowA][js + half * 16 + e];
        aP[8 + e] = p_s[rowA][js + half * 16 + 8 + e];
      }
      v16h bV = *(const v16h*)(vbase + (size_t)rowA * kS + j0 + js + half * 16);
      O = wmma_f16(aP, bV, O);
    }
  }

  if (half == 0) sum_s[rowA] = ssum;
  __syncthreads();
#pragma unroll
  for (int v = 0; v < 8; ++v) {
    int M = v + 8 * half;
    float o = O[v] / sum_s[M];
    avec[(((size_t)(i0 + M) * kB + b) * kNH + n) * kDH + rowA] = (_Float16)o;
  }
}

// ---------------------------------------------------------------------------
// Output projection (K=192 over n,d) + residual + LayerNorm; 1 wave / 16 toks
// ---------------------------------------------------------------------------
__global__ __launch_bounds__(32)
void k_out(const _Float16* __restrict__ avec, const float* __restrict__ Wo,
           const float* __restrict__ x, const float* __restrict__ lnw,
           const float* __restrict__ lnb, float* __restrict__ out) {
  __shared__ float o_s[16][kDM];
  const int t0 = blockIdx.x * 16;
  const int l = threadIdx.x;
  const int rowA = l & 15, kbA = (l >> 4) * 8;
  const int half = l >> 4;
  v8f acc[12] = {};
  for (int kk = 0; kk < kDM; kk += 32) {
    v16h a = {};
    const _Float16* ap = avec + (size_t)(t0 + rowA) * kDM + kk + kbA;
#pragma unroll
    for (int e = 0; e < 8; ++e) { a[e] = ap[e]; a[8 + e] = ap[16 + e]; }
#pragma unroll
    for (int hb = 0; hb < 12; ++hb) {
      v16h bW;
      const float* wp = Wo + (size_t)(hb * 16 + rowA) * kDM + kk + half * 8;
#pragma unroll
      for (int e = 0; e < 8; ++e) { bW[e] = (_Float16)wp[e]; bW[8 + e] = (_Float16)wp[16 + e]; }
      acc[hb] = wmma_f16(a, bW, acc[hb]);
    }
  }
#pragma unroll
  for (int hb = 0; hb < 12; ++hb) {
#pragma unroll
    for (int v = 0; v < 8; ++v) {
      int M = v + 8 * half;
      int h = hb * 16 + rowA;
      o_s[M][h] = acc[hb][v] + x[(size_t)(t0 + M) * kDM + h];  // + residual
    }
  }
  __syncthreads();
  float s1 = 0.0f, s2 = 0.0f;
  const float* rp = &o_s[rowA][half * 96];
#pragma unroll 8
  for (int e = 0; e < 96; ++e) { float v = rp[e]; s1 += v; s2 += v * v; }
  s1 += __shfl_xor(s1, 16, 32);
  s2 += __shfl_xor(s2, 16, 32);
  float mean = s1 * (1.0f / kDM);
  float var  = s2 * (1.0f / kDM) - mean * mean;
  float inv  = rsqrtf(var + 1e-12f);
  float* op = out + (size_t)(t0 + rowA) * kDM + half * 96;
  for (int e = 0; e < 96; ++e) {
    int h = half * 96 + e;
    op[e] = (rp[e] - mean) * inv * lnw[h] + lnb[h];
  }
}

// ---------------------------------------------------------------------------
extern "C" void kernel_launch(void* const* d_in, const int* in_sizes, int n_in,
                              void* d_out, int out_size, void* d_ws, size_t ws_size,
                              hipStream_t stream) {
  (void)in_sizes; (void)n_in; (void)out_size; (void)ws_size;
  const float* x   = (const float*)d_in[0];
  const float* r   = (const float*)d_in[1];
  // d_in[2] = mask0: reference zeroes it -> unused
  const float* Wq  = (const float*)d_in[3];
  const float* Wk  = (const float*)d_in[4];
  const float* Wv  = (const float*)d_in[5];
  const float* Wo  = (const float*)d_in[6];
  const float* Wr  = (const float*)d_in[7];
  const float* rwb = (const float*)d_in[8];
  const float* rrb = (const float*)d_in[9];
  const float* lnw = (const float*)d_in[10];
  const float* lnb = (const float*)d_in[11];
  float* out = (float*)d_out;

  char* ws = (char*)d_ws;
  size_t off = 0;
  auto carve = [&](size_t bytes) -> char* {
    char* p = ws + off;
    off += (bytes + 255) & ~(size_t)255;
    return p;
  };
  _Float16* qw   = (_Float16*)carve((size_t)kBN * kS * kDH * sizeof(_Float16));
  _Float16* qr   = (_Float16*)carve((size_t)kBN * kS * kDH * sizeof(_Float16));
  _Float16* kmat = (_Float16*)carve((size_t)kBN * kS * kDH * sizeof(_Float16));
  _Float16* vT   = (_Float16*)carve((size_t)kBN * kS * kDH * sizeof(_Float16));
  _Float16* krm  = (_Float16*)carve((size_t)kBN * kRP * kDH * sizeof(_Float16));
  _Float16* avec = (_Float16*)carve((size_t)kS * kB * kDM * sizeof(_Float16));

  k_proj_x<<<dim3(kBN, kS / 64), 128, 0, stream>>>(x, Wq, Wk, Wv, rwb, rrb,
                                                   qw, qr, kmat, vT);
  k_proj_r<<<dim3(kBN, kR / 64), 128, 0, stream>>>(r, Wr, krm);
  k_pad_kr<<<(kBN * 16 * kDH + 255) / 256, 256, 0, stream>>>(krm);
  k_attn<<<kBN * (kS / 16), 32, 0, stream>>>(qw, qr, kmat, krm, vT, avec);
  k_out<<<(kS * kB) / 16, 32, 0, stream>>>(avec, Wo, x, lnw, lnb, out);
}
